// SelfAttention_38414187495438
// MI455X (gfx1250) — matrix operand order repaired and checked by reading, not compiled, not run on hardware
//
#include <hip/hip_runtime.h>

// ---------------- problem constants ----------------
#define BB     2
#define CC     512
#define LL     4096
#define HH     8
#define DD     64      // CC / HH

// ---------------- WMMA vector types ----------------
typedef __attribute__((ext_vector_type(16))) __bf16 v16bf;
typedef __attribute__((ext_vector_type(8)))  float  v8f;

union BF16x16 { v16bf v; uint4 q[2]; unsigned short u[16]; };
union F32x8   { v8f  v; float f[8]; };

__device__ __forceinline__ unsigned short f32_to_bf16(float f) {
  unsigned int u = __float_as_uint(f);
  unsigned int r = u + 0x7fffu + ((u >> 16) & 1u);   // round-to-nearest-even
  return (unsigned short)(r >> 16);
}

// ---------------- plain f32 -> bf16 ----------------
__global__ void cvt_bf16_kernel(const float* __restrict__ in,
                                unsigned short* __restrict__ out, int n) {
  int i = blockIdx.x * blockDim.x + threadIdx.x;
  if (i < n) out[i] = f32_to_bf16(in[i]);
}

// ---------------- tiled transpose + convert: [B][C][L] f32 -> [B][L][C] bf16 ----
__global__ __launch_bounds__(256) void cvt_transpose_kernel(
    const float* __restrict__ in, unsigned short* __restrict__ outT) {
  __shared__ float T[32][33];
  const int b  = blockIdx.z;
  const int c0 = blockIdx.y * 32, l0 = blockIdx.x * 32;
  const int tx = threadIdx.x & 31, ty = threadIdx.x >> 5;   // 32 x 8
  for (int i = 0; i < 4; ++i) {
    const int c = ty + i * 8;
    T[c][tx] = in[((size_t)b * CC + c0 + c) * LL + l0 + tx];
  }
  __syncthreads();
  for (int i = 0; i < 4; ++i) {
    const int l = ty + i * 8;
    outT[((size_t)b * LL + l0 + l) * CC + c0 + tx] = f32_to_bf16(T[tx][l]);
  }
}

// ---------------- projection GEMM ----------------
// O[o,l] = sum_c W[o,c] * Xt[b,l,c].  o spans {Wq->Q, Wm[0:512]->K, Wm[512:1024]->V}.
// Q,K stored [B*H][L][64]; V stored transposed [B*H][64][L] for the PV fragment feed.
__global__ __launch_bounds__(256) void proj_kernel(
    const unsigned short* __restrict__ Xt,    // [B][L][C] bf16
    const unsigned short* __restrict__ Wqbf,  // [512][512]  bf16
    const unsigned short* __restrict__ Wmbf,  // [1024][512] bf16
    unsigned short* __restrict__ Qbf,
    unsigned short* __restrict__ Kbf,
    unsigned short* __restrict__ Vt) {
  __shared__ uint4 Wlds[2][128 * 5];   // 128 rows x 32 bf16 (4 uint4) + 1 pad
  __shared__ uint4 Xlds[2][64 * 5];    // 64 rows  x 32 bf16 (4 uint4) + 1 pad

  const int tid  = threadIdx.x;
  const int lane = tid & 31;
  const int wave = tid >> 5;
  const int half = lane >> 4;
  const int l16  = lane & 15;

  const int b     = blockIdx.z;
  const int lbase = blockIdx.x * 64;
  const int oglob = blockIdx.y * 128;       // 0..1535
  const int seg   = oglob >> 9;             // 0:Q  1:K  2:V
  const int oseg  = oglob & 511;

  const unsigned short* Wblk = (seg == 0) ? (Wqbf + (size_t)oglob * CC)
                                          : (Wmbf + (size_t)(oglob - 512) * CC);
  const float scale = (seg == 0) ? 0.125f : 1.0f;   // fold D^-0.5 into Q

  const uint4* Wsrc4 = (const uint4*)Wblk;                                  // row stride 64 uint4
  const uint4* Xsrc4 = (const uint4*)Xt + ((size_t)b * LL + lbase) * (CC / 8); // row stride 64 uint4

  // cooperative-load index precompute
  const int wr0 = (tid + 0)   >> 2, wj0 = (tid + 0)   & 3;   // W: 512 uint4, 2/thread
  const int wr1 = (tid + 256) >> 2, wj1 = (tid + 256) & 3;
  const int xr  = tid >> 2,         xj  = tid & 3;           // X: 256 uint4, 1/thread

  uint4 wreg0, wreg1, xreg;
  auto load_regs = [&](int ck) {
    wreg0 = Wsrc4[(size_t)wr0 * 64 + ck * 4 + wj0];
    wreg1 = Wsrc4[(size_t)wr1 * 64 + ck * 4 + wj1];
    xreg  = Xsrc4[(size_t)xr  * 64 + ck * 4 + xj];
  };
  auto store_lds = [&](int buf) {
    Wlds[buf][wr0 * 5 + wj0] = wreg0;
    Wlds[buf][wr1 * 5 + wj1] = wreg1;
    Xlds[buf][xr  * 5 + xj]  = xreg;
  };

  F32x8 acc[4];
  for (int s = 0; s < 4; ++s)
    for (int r = 0; r < 8; ++r) acc[s].f[r] = 0.f;

  const int wrow = wave * 16 + l16;

  load_regs(0);
  store_lds(0);
  int cur = 0;
  for (int ck = 0; ck < CC / 32; ++ck) {
    if (ck + 1 < CC / 32) load_regs(ck + 1);   // in flight during compute
    __syncthreads();
    const uint4* Wb = &Wlds[cur][0];
    const uint4* Xb = &Xlds[cur][0];
    BF16x16 aw;                                 // A: rows=o, interleaved k runs
    aw.q[0] = Wb[wrow * 5 + half];
    aw.q[1] = Wb[wrow * 5 + half + 2];
    for (int sub = 0; sub < 4; ++sub) {         // B: lane=l col, interleaved k runs
      BF16x16 bx;
      const int xl = sub * 16 + l16;
      bx.q[0] = Xb[xl * 5 + half];
      bx.q[1] = Xb[xl * 5 + half + 2];
      acc[sub].v = __builtin_amdgcn_wmma_f32_16x16x32_bf16(
          false, aw.v, false, bx.v, (short)0, acc[sub].v, false, false);
    }
    if (ck + 1 < CC / 32) store_lds(1 - cur);
    cur ^= 1;
  }

  for (int r = 0; r < 8; ++r) {
    const int os = oseg + wave * 16 + r + 8 * half;
    const int h = os >> 6, d = os & 63;
    for (int sub = 0; sub < 4; ++sub) {
      const int l = lbase + sub * 16 + l16;
      const unsigned short v = f32_to_bf16(acc[sub].f[r] * scale);
      if (seg == 2)
        Vt[(((size_t)b * HH + h) * DD + d) * LL + l] = v;            // [BH][64][L]
      else if (seg == 1)
        Kbf[(((size_t)b * HH + h) * LL + l) * DD + d] = v;           // [BH][L][64]
      else
        Qbf[(((size_t)b * HH + h) * LL + l) * DD + d] = v;
    }
  }
}

// ---------------- flash attention ----------------
// grid: (L/128, B*H); 256 threads = 8 waves; wave owns 16 query rows; key tile = 64.
__global__ __launch_bounds__(256) void attn_kernel(
    const unsigned short* __restrict__ Qbf,   // [B*H][L][64] bf16 (pre-scaled)
    const unsigned short* __restrict__ Kbf,   // [B*H][L][64]
    const unsigned short* __restrict__ VtT,   // [B*H][64][L]
    const int* __restrict__ mask,             // [B][L]
    float* __restrict__ out) {                // [B][C][L] f32
  __shared__ uint4 Klds[2][64 * 9];                        // 64 rows x 64 bf16 (8 uint4) + pad
  __shared__ uint4 Vlds[2][64 * 9];                        // 64 d    x 64 k    (8 uint4) + pad
  __shared__ __align__(16) unsigned short Plds[8][16 * 72]; // per-wave 16 x 64 + pad

  const int tid  = threadIdx.x;
  const int lane = tid & 31;
  const int wave = tid >> 5;
  const int half = lane >> 4;
  const int l16  = lane & 15;

  const int bh = blockIdx.y;
  const int b  = bh >> 3;
  const int q0 = blockIdx.x * 128 + wave * 16;

  // ---- Q A-fragments (global_load_b128 x4, once)
  BF16x16 aq[2];
  {
    const uint4* qrow4 = (const uint4*)(Qbf + ((size_t)bh * LL + q0 + l16) * DD);
    for (int ch = 0; ch < 2; ++ch) {
      aq[ch].q[0] = qrow4[ch * 4 + half];
      aq[ch].q[1] = qrow4[ch * 4 + half + 2];
    }
  }

  F32x8 acc[4];
  float mrow[8], lrow[8];
  for (int s = 0; s < 4; ++s)
    for (int r = 0; r < 8; ++r) acc[s].f[r] = 0.f;
  for (int r = 0; r < 8; ++r) { mrow[r] = -INFINITY; lrow[r] = 0.f; }

  const uint4* Ksrc4 = (const uint4*)Kbf + (size_t)bh * LL * 8;          // flat tiles
  const uint4* Vt4   = (const uint4*)VtT + (size_t)bh * DD * (LL / 8);   // row stride 512 uint4
  const int* maskrow = mask + b * LL;
  unsigned short* myP = &Plds[wave][0];
  const uint4* P4 = (const uint4*)myP;

  // cooperative-load index precompute: 512 uint4 per tile per matrix, 2/thread
  const int r0 = (tid + 0)   >> 3, j0 = (tid + 0)   & 7;
  const int r1 = (tid + 256) >> 3, j1 = (tid + 256) & 7;

  uint4 kreg0, kreg1, vreg0, vreg1;
  auto load_regs = [&](int kt) {
    const size_t kt512 = (size_t)kt * 512;
    kreg0 = Ksrc4[kt512 + tid];
    kreg1 = Ksrc4[kt512 + tid + 256];
    vreg0 = Vt4[(size_t)r0 * 512 + kt * 8 + j0];
    vreg1 = Vt4[(size_t)r1 * 512 + kt * 8 + j1];
  };
  auto store_lds = [&](int buf) {
    Klds[buf][r0 * 9 + j0] = kreg0;
    Klds[buf][r1 * 9 + j1] = kreg1;
    Vlds[buf][r0 * 9 + j0] = vreg0;
    Vlds[buf][r1 * 9 + j1] = vreg1;
  };

  const int NT = LL / 64;
  load_regs(0);
  store_lds(0);
  int cur = 0;

  for (int kt = 0; kt < NT; ++kt) {
    const int kbase = kt * 64;
    if (kt + 1 < NT) load_regs(kt + 1);       // global loads overlap compute
    __syncthreads();
    const uint4* Kb = &Klds[cur][0];
    const uint4* Vb = &Vlds[cur][0];

    // ---- S = Q K^T : 4 n-subtiles x 2 d-chunks = 8 WMMAs
    F32x8 s[4];
    for (int ns = 0; ns < 4; ++ns) {
      v8f z = {};
      s[ns].v = z;
      const int krow = ns * 16 + l16;
      for (int ch = 0; ch < 2; ++ch) {
        BF16x16 bk;
        bk.q[0] = Kb[krow * 9 + ch * 4 + half];
        bk.q[1] = Kb[krow * 9 + ch * 4 + half + 2];
        s[ns].v = __builtin_amdgcn_wmma_f32_16x16x32_bf16(
            false, aq[ch].v, false, bk.v, (short)0, s[ns].v, false, false);
      }
    }

    // ---- mask bias (column-wise)
    float bias[4];
    for (int ks = 0; ks < 4; ++ks)
      bias[ks] = maskrow[kbase + ks * 16 + l16] ? 0.f : -1e30f;

    // ---- online softmax: rows r+8*half reduce across the 16 lanes of each half
    for (int r = 0; r < 8; ++r) {
      const float s0 = s[0].f[r] + bias[0];
      const float s1 = s[1].f[r] + bias[1];
      const float s2 = s[2].f[r] + bias[2];
      const float s3 = s[3].f[r] + bias[3];
      float v = fmaxf(fmaxf(s0, s1), fmaxf(s2, s3));
      for (int off = 1; off < 16; off <<= 1)
        v = fmaxf(v, __shfl_xor(v, off, 32));
      const float mnew = fmaxf(fmaxf(mrow[r], v), -1e30f);
      const float corr = __expf(mrow[r] - mnew);
      const float p0 = __expf(s0 - mnew);
      const float p1 = __expf(s1 - mnew);
      const float p2 = __expf(s2 - mnew);
      const float p3 = __expf(s3 - mnew);
      float ps = (p0 + p1) + (p2 + p3);
      for (int off = 1; off < 16; off <<= 1)
        ps += __shfl_xor(ps, off, 32);
      lrow[r] = lrow[r] * corr + ps;
      mrow[r] = mnew;
      for (int sub = 0; sub < 4; ++sub) acc[sub].f[r] *= corr;
      const int prow = r + 8 * half;
      myP[prow * 72 + l16]      = f32_to_bf16(p0);
      myP[prow * 72 + 16 + l16] = f32_to_bf16(p1);
      myP[prow * 72 + 32 + l16] = f32_to_bf16(p2);
      myP[prow * 72 + 48 + l16] = f32_to_bf16(p3);
    }

    // ---- O += P V : 2 k-chunks x 4 d-subtiles = 8 WMMAs
    for (int kc = 0; kc < 2; ++kc) {
      BF16x16 ap;                                    // A-layout reload of P
      ap.q[0] = P4[l16 * 9 + kc * 4 + half];
      ap.q[1] = P4[l16 * 9 + kc * 4 + half + 2];
      for (int sub = 0; sub < 4; ++sub) {
        BF16x16 bv;                                  // Vt row = d, interleaved k runs
        const int vd = sub * 16 + l16;
        bv.q[0] = Vb[vd * 9 + kc * 4 + half];
        bv.q[1] = Vb[vd * 9 + kc * 4 + half + 2];
        acc[sub].v = __builtin_amdgcn_wmma_f32_16x16x32_bf16(
            false, ap.v, false, bv.v, (short)0, acc[sub].v, false, false);
      }
    }

    if (kt + 1 < NT) store_lds(1 - cur);
    cur ^= 1;
  }

  // ---- epilogue: normalize and scatter to [B][C][L] f32
  for (int r = 0; r < 8; ++r) {
    const float inv = lrow[r] > 0.f ? 1.f / lrow[r] : 0.f;
    const int lq = q0 + r + 8 * half;
    for (int sub = 0; sub < 4; ++sub) {
      const int c = (bh & 7) * DD + sub * 16 + l16;
      out[((size_t)b * CC + c) * LL + lq] = acc[sub].f[r] * inv;
    }
  }
}

// ---------------- host launcher ----------------
extern "C" void kernel_launch(void* const* d_in, const int* in_sizes, int n_in,
                              void* d_out, int out_size, void* d_ws, size_t ws_size,
                              hipStream_t stream) {
  const float* queries = (const float*)d_in[0];   // [B, C, L]
  const int*   mask    = (const int*)d_in[1];     // [B, L]
  const float* Wm      = (const float*)d_in[2];   // [2C, C]
  const float* Wq      = (const float*)d_in[3];   // [C, C]
  float* out = (float*)d_out;

  char* ws = (char*)d_ws;
  const size_t nX  = (size_t)BB * CC * LL;
  const size_t nWq = (size_t)CC * CC;
  const size_t nWm = (size_t)2 * CC * CC;

  unsigned short* Xt   = (unsigned short*)ws;   ws += nX  * 2;  // [B][L][C]
  unsigned short* Wqbf = (unsigned short*)ws;   ws += nWq * 2;
  unsigned short* Wmbf = (unsigned short*)ws;   ws += nWm * 2;
  unsigned short* Qbf  = (unsigned short*)ws;   ws += nX  * 2;  // [BH][L][64]
  unsigned short* Kbf  = (unsigned short*)ws;   ws += nX  * 2;  // [BH][L][64]
  unsigned short* Vt   = (unsigned short*)ws;   ws += nX  * 2;  // [BH][64][L]

  cvt_transpose_kernel<<<dim3(LL / 32, CC / 32, BB), 256, 0, stream>>>(queries, Xt);
  cvt_bf16_kernel<<<(int)((nWq + 255) / 256), 256, 0, stream>>>(Wq, Wqbf, (int)nWq);
  cvt_bf16_kernel<<<(int)((nWm + 255) / 256), 256, 0, stream>>>(Wm, Wmbf, (int)nWm);

  proj_kernel<<<dim3(LL / 64, 12, BB), 256, 0, stream>>>(Xt, Wqbf, Wmbf, Qbf, Kbf, Vt);
  attn_kernel<<<dim3(LL / 128, BB * HH), 256, 0, stream>>>(Qbf, Kbf, Vt, mask, out);
}
